// DDSP_signal_only_1898375544959
// MI455X (gfx1250) — matrix operand order repaired and checked by reading, not compiled, not run on hardware
//
#include <hip/hip_runtime.h>
#include <hip/hip_bf16.h>
#include <math.h>

typedef _Float16 f16;
typedef _Float16 v16h __attribute__((ext_vector_type(16)));
typedef _Float16 h2   __attribute__((ext_vector_type(2)));
typedef float    v8f  __attribute__((ext_vector_type(8)));

#define SRATE   16000
#define HSZ     512
#define NHARM   100
#define NBANDS  65
#define NBATCH  4
#define NFRAME  1000
#define NTOK    (NBATCH*NFRAME)      // 4000 tokens, rows ordered b*NFRAME+f
#define LSIG    160000
#define PI_F    3.14159265358979f
#define LOG10_F 2.302585092994046f

// ---------------------------------------------------------------------------
// f16 WMMA GEMM, 4 N-tiles per wave (A-fragment reuse x4), branch-free inner
// loop. REQUIRES Npad % 64 == 0 (B buffer zero-padded to 4-tile groups).
//   C[MxN](f32) = A[MxKp](f16,row-major) @ Wt[NpadxKp](f16) + bias
// ---------------------------------------------------------------------------
__global__ void k_gemm(float* __restrict__ C, const f16* __restrict__ A,
                       const f16* __restrict__ Bt_, const float* __restrict__ bias,
                       int M, int N, int Npad, int Kp) {
  int wave = (int)((blockIdx.x * blockDim.x + threadIdx.x) >> 5);
  int lane = threadIdx.x & 31;
  int ngroups = Npad >> 6;                        // 4 tiles (64 cols) per wave
  int tm = wave / ngroups, tg = wave - tm * ngroups;
  if (tm * 16 >= M) return;                       // wave-uniform exit
  int half = lane >> 4, mn = lane & 15;
  int row = tm * 16 + mn;                         // A row (both lane halves: M=0..15)
  const f16* arow = A + (size_t)row * Kp + half * 16;
  int tn0 = tg * 4;
  const f16* brow[4];
#pragma unroll
  for (int u = 0; u < 4; ++u)
    brow[u] = Bt_ + (size_t)((tn0 + u) * 16 + mn) * Kp + half * 16;
  v8f acc[4] = {};
  for (int k0 = 0; k0 < Kp; k0 += 32) {
    v16h af;
#pragma unroll
    for (int v = 0; v < 4; ++v) {                 // A: vgpr v -> K=2v+16h ; v+4 -> K=8+2v+16h
      h2 lo = *(const h2*)(arow + k0 + 2 * v);
      h2 hi = *(const h2*)(arow + k0 + 8 + 2 * v);
      af[2 * v] = lo[0]; af[2 * v + 1] = lo[1];
      af[2 * v + 8] = hi[0]; af[2 * v + 9] = hi[1];
    }
#pragma unroll
    for (int u = 0; u < 4; ++u) {
      v16h bf;
#pragma unroll
      for (int v = 0; v < 8; ++v) {               // B: vgpr v -> K=2v+16h at column n
        h2 bb = *(const h2*)(brow[u] + k0 + 2 * v);
        bf[2 * v] = bb[0]; bf[2 * v + 1] = bb[1];
      }
      acc[u] = __builtin_amdgcn_wmma_f32_16x16x32_f16(false, af, false, bf,
                                                      (short)0, acc[u], false, false);
    }
  }
#pragma unroll
  for (int u = 0; u < 4; ++u) {
    int col = (tn0 + u) * 16 + mn;
    float bv = (col < N && bias) ? bias[col] : 0.f;
#pragma unroll
    for (int r = 0; r < 8; ++r) {                 // C: vgpr r -> M = r + 8*half
      int rr = tm * 16 + r + 8 * half;
      if (rr < M && col < N) C[(size_t)rr * N + col] = acc[u][r] + bv;
    }
  }
}

// ---------------------------------------------------------------------------
// LayerNorm(512) + LeakyReLU, f32 in -> f16 out (strided dest for in-place concat)
// ---------------------------------------------------------------------------
__global__ void k_ln_leaky(f16* __restrict__ dst, const float* __restrict__ Y,
                           const float* __restrict__ g, const float* __restrict__ be,
                           int dstStride, int colOff) {
  __shared__ float red[256];
  int row = blockIdx.x, t = threadIdx.x;
  const float* y = Y + (size_t)row * HSZ;
  float v0 = y[t], v1 = y[t + 256];
  red[t] = v0 + v1; __syncthreads();
#pragma unroll
  for (int o = 128; o > 0; o >>= 1) { if (t < o) red[t] += red[t + o]; __syncthreads(); }
  float mean = red[0] * (1.f / 512.f); __syncthreads();
  float d0 = v0 - mean, d1 = v1 - mean;
  red[t] = d0 * d0 + d1 * d1; __syncthreads();
#pragma unroll
  for (int o = 128; o > 0; o >>= 1) { if (t < o) red[t] += red[t + o]; __syncthreads(); }
  float rs = rsqrtf(red[0] * (1.f / 512.f) + 1e-5f);
  f16* drow = dst + (size_t)row * dstStride + colOff;
  float x0 = d0 * rs * g[t] + be[t];             x0 = x0 >= 0.f ? x0 : 0.01f * x0;
  float x1 = d1 * rs * g[t + 256] + be[t + 256]; x1 = x1 >= 0.f ? x1 : 0.01f * x1;
  drow[t] = (f16)x0; drow[t + 256] = (f16)x1;
}

// Layer-1 (K=1) outer product: Y[r][n] = in[r]*W[n] + b[n]
__global__ void k_outer(float* __restrict__ Y, const float* __restrict__ inv,
                        const float* __restrict__ W, const float* __restrict__ b, int rows) {
  int i = blockIdx.x * blockDim.x + threadIdx.x;
  if (i >= rows * HSZ) return;
  int r = i >> 9, n = i & 511;
  Y[i] = inv[r] * W[n] + b[n];
}

// ---------------- weight prep ----------------
__global__ void k_w_transpose_pad(f16* __restrict__ dst, const float* __restrict__ src,
                                  int K, int N, int Kp, int Np) {
  int i = blockIdx.x * blockDim.x + threadIdx.x;
  if (i >= Np * Kp) return;
  int n = i / Kp, k = i - n * Kp;
  dst[i] = (f16)((n < N && k < K) ? src[(size_t)k * N + n] : 0.f);
}
__global__ void k_w_convert(f16* __restrict__ dst, const float* __restrict__ src, int count) {
  int i = blockIdx.x * blockDim.x + threadIdx.x;
  if (i < count) dst[i] = (f16)src[i];
}
__global__ void k_fill_f16(f16* __restrict__ dst, int count) {
  int i = blockIdx.x * blockDim.x + threadIdx.x;
  if (i < count) dst[i] = (f16)0.f;
}
__global__ void k_set_pl(f16* __restrict__ X1, const float* __restrict__ pitch,
                         const float* __restrict__ loud) {
  int r = blockIdx.x * blockDim.x + threadIdx.x;
  if (r < NTOK) {
    X1[(size_t)r * 544 + 512] = (f16)pitch[r];
    X1[(size_t)r * 544 + 513] = (f16)loud[r];
  }
}

// ---------------------------------------------------------------------------
// GRU recurrence: persistent single workgroup (512 thr = 16 waves).
// Gi = x@Wih^T + bih precomputed. h in LDS (f16). Each wave owns 6 N-tiles;
// A fragment built once per k-step and reused for all 6 tiles (6 accumulators).
// ---------------------------------------------------------------------------
__global__ void k_gru(f16* __restrict__ gout /*stride 544*/, const float* __restrict__ Gi,
                      const f16* __restrict__ WhhT, const float* __restrict__ bhh) {
  __shared__ f16   hH[NBATCH * HSZ];
  __shared__ float hF[NBATCH * HSZ];
  __shared__ float Sh[NBATCH * 3 * HSZ];
  int tid = threadIdx.x;
  int wave = tid >> 5, lane = tid & 31;
  int half = lane >> 4, mn = lane & 15;
  for (int i = tid; i < NBATCH * HSZ; i += 512) { hH[i] = (f16)0.f; hF[i] = 0.f; }
  __syncthreads();
  for (int f = 0; f < NFRAME; ++f) {
    v8f acc[6] = {};
#pragma unroll 1
    for (int k0 = 0; k0 < HSZ; k0 += 32) {
      v16h af;
#pragma unroll
      for (int v = 0; v < 4; ++v) {
        int k = k0 + 16 * half + 2 * v;
        f16 a0 = (f16)0.f, a1 = (f16)0.f, a2 = (f16)0.f, a3 = (f16)0.f;
        if (mn < NBATCH) {
          a0 = hH[mn * HSZ + k];      a1 = hH[mn * HSZ + k + 1];
          a2 = hH[mn * HSZ + k + 8]; a3 = hH[mn * HSZ + k + 9];
        }
        af[2 * v] = a0; af[2 * v + 1] = a1; af[2 * v + 8] = a2; af[2 * v + 9] = a3;
      }
#pragma unroll
      for (int tile = 0; tile < 6; ++tile) {
        int col = (wave * 6 + tile) * 16 + mn;
        const f16* brow = WhhT + (size_t)col * HSZ + half * 16 + k0;
        v16h bf;
#pragma unroll
        for (int v = 0; v < 8; ++v) {
          h2 bb = *(const h2*)(brow + 2 * v);
          bf[2 * v] = bb[0]; bf[2 * v + 1] = bb[1];
        }
        acc[tile] = __builtin_amdgcn_wmma_f32_16x16x32_f16(false, af, false, bf,
                                                           (short)0, acc[tile], false, false);
      }
    }
    if (half == 0) {
#pragma unroll
      for (int tile = 0; tile < 6; ++tile) {
        int col = (wave * 6 + tile) * 16 + mn;
#pragma unroll
        for (int r = 0; r < NBATCH; ++r) Sh[r * 3 * HSZ + col] = acc[tile][r];
      }
    }
    __syncthreads();
    // fused gates: r,z = sigmoid(gi+gh), n = tanh(i_n + r*h_n), h = (1-z)n + z h
    int j = tid;
#pragma unroll 1
    for (int b = 0; b < NBATCH; ++b) {
      size_t grow = ((size_t)b * NFRAME + f) * (3 * HSZ);
      float gr = Gi[grow + j]                  + Sh[b * 3 * HSZ + j]       + bhh[j];
      float gz = Gi[grow + HSZ + j]            + Sh[b * 3 * HSZ + HSZ + j] + bhh[HSZ + j];
      float hn = Sh[b * 3 * HSZ + 2 * HSZ + j] + bhh[2 * HSZ + j];
      float in = Gi[grow + 2 * HSZ + j];
      float r_ = 1.f / (1.f + __expf(-gr));
      float z_ = 1.f / (1.f + __expf(-gz));
      float n_ = tanhf(in + r_ * hn);
      float hnew = (1.f - z_) * n_ + z_ * hF[b * HSZ + j];
      hF[b * HSZ + j] = hnew;
      hH[b * HSZ + j] = (f16)hnew;
      gout[((size_t)b * NFRAME + f) * 544 + j] = (f16)hnew;
    }
    __syncthreads();
  }
}

// ---------------- synthesis ----------------
__global__ void k_harm_amps(float* __restrict__ amps, const float* __restrict__ Yh,
                            const float* __restrict__ pitch) {
  __shared__ float sh[128];
  __shared__ float ph[NHARM + 1];
  int row = blockIdx.x, t = threadIdx.x;
  if (t < NHARM + 1) {
    float x = Yh[(size_t)row * (NHARM + 1) + t];
    float sg = 1.f / (1.f + __expf(-x));
    ph[t] = 2.f * __powf(sg, LOG10_F) + 1e-7f;
  }
  __syncthreads();
  float p = pitch[row], a = 0.f;
  if (t >= 1 && t <= NHARM) {
    float aa = (p * (float)t < 0.5f * (float)SRATE ? 1.f : 0.f) + 1e-4f;
    a = ph[t] * aa;
  }
  sh[t] = a; __syncthreads();
#pragma unroll
  for (int o = 64; o > 0; o >>= 1) { if (t < o) sh[t] += sh[t + o]; __syncthreads(); }
  if (t >= 1 && t <= NHARM) amps[(size_t)row * NHARM + (t - 1)] = a / sh[0] * ph[0];
}

__global__ void k_noise_scale(float* __restrict__ pn, const float* __restrict__ Yn, int count) {
  int i = blockIdx.x * blockDim.x + threadIdx.x;
  if (i >= count) return;
  float x = Yn[i] - 5.f;
  float sg = 1.f / (1.f + __expf(-x));
  pn[i] = 2.f * __powf(sg, LOG10_F) + 1e-7f;
}

// per-batch prefix phase (frames are constant-increment blocks of 160)
__global__ void k_phase(float* __restrict__ C, const float* __restrict__ pitch) {
  int b = threadIdx.x;
  if (b < NBATCH) {
    float acc = 0.f;
    for (int f = 0; f < NFRAME; ++f) {
      C[b * NFRAME + f] = acc;
      acc += 160.f * 2.f * PI_F * pitch[b * NFRAME + f] / (float)SRATE;
    }
  }
}

__global__ void k_harmonic(float* __restrict__ sig, const float* __restrict__ amps,
                           const float* __restrict__ pitch, const float* __restrict__ C) {
  __shared__ float am[NHARM];
  int row = blockIdx.x, s = threadIdx.x;                // 160 threads = 1 frame
  if (s < NHARM) am[s] = amps[(size_t)row * NHARM + s];
  __syncthreads();
  float w = 2.f * PI_F * pitch[row] / (float)SRATE;
  float om = C[row] + (float)(s + 1) * w;
  float acc = 0.f;
#pragma unroll 4
  for (int k = 0; k < NHARM; ++k) acc += __sinf(om * (float)(k + 1)) * am[k];
  int b = row / NFRAME, f = row - b * NFRAME;
  sig[(size_t)b * LSIG + f * 160 + s] = acc;
}

// per-frame: irfft(65) -> windowed/rolled 160-tap kernel -> causal conv, add into sig
__global__ void k_noise(float* __restrict__ sig, const float* __restrict__ pn,
                        const float* __restrict__ noise_in) {
  __shared__ float X[NBANDS];
  __shared__ float ir128[128];
  __shared__ float fk[160];
  __shared__ float nz[160];
  int row = blockIdx.x, t = threadIdx.x;                // 160 threads
  if (t < NBANDS) X[t] = pn[(size_t)row * NBANDS + t];
  nz[t] = noise_in[(size_t)row * 160 + t];
  __syncthreads();
  if (t < 128) {
    float acc = X[0] + X[64] * __cosf(PI_F * (float)t);
    for (int k = 1; k < 64; ++k)
      acc += 2.f * X[k] * __cosf(2.f * PI_F * (float)k * (float)t * (1.f / 128.f));
    ir128[t] = acc * (1.f / 128.f);
  }
  __syncthreads();
  float v;
  if (t < 64)       v = ir128[t]      * (0.5f - 0.5f * __cosf(2.f * PI_F * (float)(t + 64) * (1.f / 128.f)));
  else if (t < 96)  v = 0.f;
  else              v = ir128[t - 32] * (0.5f - 0.5f * __cosf(2.f * PI_F * (float)(t - 96) * (1.f / 128.f)));
  fk[t] = v;
  __syncthreads();
  float acc = 0.f;
  for (int j = 0; j <= t; ++j) acc += nz[j] * fk[t - j];
  int b = row / NFRAME, f = row - b * NFRAME;
  sig[(size_t)b * LSIG + f * 160 + t] += acc;
}

// ---------------- reverb ----------------
__global__ void k_impulse(float* __restrict__ imp, const float* __restrict__ rev_noise,
                          const float* __restrict__ decay, const float* __restrict__ wet) {
  int i = blockIdx.x * blockDim.x + threadIdx.x;
  if (i >= LSIG) return;
  float sp = log1pf(__expf(-decay[0]));                 // softplus(-decay)
  float w = 1.f / (1.f + __expf(-wet[0]));
  float v = rev_noise[i] * __expf(-sp * ((float)i / (float)SRATE) * 500.f) * w;
  imp[i] = (i == 0) ? 1.f : v;
}
__global__ void k_prep_sigH(f16* __restrict__ sigH, const float* __restrict__ sig) {
  int i = blockIdx.x * blockDim.x + threadIdx.x;
  const int per = LSIG + 64;
  if (i >= NBATCH * per) return;
  int b = i / per, j = i - b * per;
  sigH[i] = (f16)(j < LSIG ? sig[(size_t)b * LSIG + j] : 0.f);
}
__global__ void k_prep_R(f16* __restrict__ R, const float* __restrict__ imp) {
  int x = blockIdx.x * blockDim.x + threadIdx.x;
  if (x >= LSIG + 128) return;
  int d = LSIG + 32 - x;                                // R[x] = imp[LSIG+32-x], reversed+guarded
  R[x] = (f16)((d >= 0 && d < LSIG) ? imp[d] : 0.f);
}
// out[b][t] = sum_j sig[b][j]*imp[t-j] as Toeplitz WMMA.
// A fragment depends only on lag j0, NOT on t -> one wave computes 4 consecutive
// t-tiles (64 samples x 4 batches) per A fragment. Branch-free inner loop:
// the reversed-impulse zero guard absorbs the j>t corner (all lags negative
// there map into the zero guard band), so no triangular conditionals needed.
__global__ void k_reverb(float* __restrict__ out, const f16* __restrict__ sigH,
                         const f16* __restrict__ R) {
  int wave = (int)((blockIdx.x * blockDim.x + threadIdx.x) >> 5);
  int lane = threadIdx.x & 31;
  if (wave >= LSIG / 64) return;                        // wave-uniform
  int t0 = wave * 64;
  int half = lane >> 4, mn = lane & 15;
  const int per = LSIG + 64;
  const f16* arow = sigH + (size_t)(mn < NBATCH ? mn : 0) * per;
  const f16* rbase[4];
#pragma unroll
  for (int u = 0; u < 4; ++u)
    rbase[u] = R + (LSIG + 32 - (t0 + u * 16 + mn) + 16 * half);
  v8f acc[4] = {};
  for (int j0 = 0; j0 < t0 + 64; j0 += 32) {
    v16h af = {};
    if (mn < NBATCH) {
      const f16* ap = arow + j0 + 16 * half;
      __builtin_prefetch(ap + 64, 0, 1);
#pragma unroll
      for (int v = 0; v < 4; ++v) {
        h2 lo = *(const h2*)(ap + 2 * v);
        h2 hi = *(const h2*)(ap + 8 + 2 * v);
        af[2 * v] = lo[0]; af[2 * v + 1] = lo[1];
        af[2 * v + 8] = hi[0]; af[2 * v + 9] = hi[1];
      }
    }
#pragma unroll
    for (int u = 0; u < 4; ++u) {
      const f16* rp = rbase[u] + j0;
      v16h bf;
#pragma unroll
      for (int v = 0; v < 8; ++v) { bf[2 * v] = rp[2 * v]; bf[2 * v + 1] = rp[2 * v + 1]; }
      acc[u] = __builtin_amdgcn_wmma_f32_16x16x32_f16(false, af, false, bf,
                                                      (short)0, acc[u], false, false);
    }
  }
  if (half == 0) {
#pragma unroll
    for (int u = 0; u < 4; ++u)
#pragma unroll
      for (int r = 0; r < NBATCH; ++r)
        out[(size_t)r * LSIG + t0 + u * 16 + mn] = acc[u][r];
  }
}

// ---------------------------------------------------------------------------
// Host side
// ---------------------------------------------------------------------------
static inline int ceil_div(long a, long b) { return (int)((a + b - 1) / b); }

static void launch_gemm(hipStream_t s, float* C, const f16* A, const f16* Bt,
                        const float* bias, int M, int N, int Npad, int Kp) {
  long ngroups = Npad / 64;                             // Npad must be mult of 64
  long waves = (long)(M / 16) * ngroups;
  k_gemm<<<ceil_div(waves * 32, 256), 256, 0, s>>>(C, A, Bt, bias, M, N, Npad, Kp);
}

extern "C" void kernel_launch(void* const* d_in, const int* in_sizes, int n_in,
                              void* d_out, int out_size, void* d_ws, size_t ws_size,
                              hipStream_t stream) {
  (void)in_sizes; (void)n_in; (void)out_size; (void)ws_size;
  // inputs in setup_inputs() dict order (depth-first, insertion order)
  const float* pitch    = (const float*)d_in[0];
  const float* loud     = (const float*)d_in[1];
  const float* noise_in = (const float*)d_in[2];
  const float *iW[3], *ib[3], *ig[3], *ibe[3], *oW[3], *ob[3], *og[3], *obe[3];
  for (int l = 0; l < 3; ++l) {
    iW[l] = (const float*)d_in[3 + 4 * l];  ib[l]  = (const float*)d_in[4 + 4 * l];
    ig[l] = (const float*)d_in[5 + 4 * l];  ibe[l] = (const float*)d_in[6 + 4 * l];
    oW[l] = (const float*)d_in[15 + 4 * l]; ob[l]  = (const float*)d_in[16 + 4 * l];
    og[l] = (const float*)d_in[17 + 4 * l]; obe[l] = (const float*)d_in[18 + 4 * l];
  }
  const float* Wih = (const float*)d_in[27]; const float* Whh = (const float*)d_in[28];
  const float* bih = (const float*)d_in[29]; const float* bhh = (const float*)d_in[30];
  const float* Wh  = (const float*)d_in[31]; const float* bh  = (const float*)d_in[32];
  const float* Wn  = (const float*)d_in[33]; const float* bn  = (const float*)d_in[34];
  const float* rev_noise = (const float*)d_in[35];
  const float* rev_decay = (const float*)d_in[36];
  const float* rev_wet   = (const float*)d_in[37];
  float* out = (float*)d_out;

  // bump allocator over d_ws
  char* ws = (char*)d_ws; size_t off = 0;
  auto alloc = [&](size_t bytes) -> void* {
    void* p = ws + off; off = (off + bytes + 255) & ~(size_t)255; return p;
  };
  f16* W2t  = (f16*)alloc(512 * 512 * 2);      // in_mlp L2^T
  f16* W3t  = (f16*)alloc(512 * 512 * 2);      // in_mlp L3^T
  f16* O1t  = (f16*)alloc(512 * 544 * 2);      // out_mlp L1^T (K 514->544 pad)
  f16* O2t  = (f16*)alloc(512 * 512 * 2);
  f16* O3t  = (f16*)alloc(512 * 512 * 2);
  f16* Wht  = (f16*)alloc(128 * 512 * 2);      // harm head^T (N 101->128, 4-tile pad)
  f16* Wnt  = (f16*)alloc(128 * 512 * 2);      // noise head^T (N 65->128, 4-tile pad)
  f16* WihC = (f16*)alloc((size_t)1536 * 1024 * 2);  // already N x K
  f16* WhhC = (f16*)alloc((size_t)1536 * 512 * 2);
  f16* Xcat = (f16*)alloc((size_t)NTOK * 1024 * 2);
  f16* Xa   = (f16*)alloc((size_t)NTOK * 512 * 2);
  f16* Xb   = (f16*)alloc((size_t)NTOK * 512 * 2);
  f16* X1   = (f16*)alloc((size_t)NTOK * 544 * 2);
  f16* sigH = (f16*)alloc((size_t)NBATCH * (LSIG + 64) * 2);
  f16* Rv   = (f16*)alloc((size_t)(LSIG + 128) * 2);
  float* Y    = (float*)alloc((size_t)NTOK * 512 * 4);
  float* Gi   = (float*)alloc((size_t)NTOK * 1536 * 4);
  float* Yh   = (float*)alloc((size_t)NTOK * (NHARM + 1) * 4);
  float* Yn   = (float*)alloc((size_t)NTOK * NBANDS * 4);
  float* amps = (float*)alloc((size_t)NTOK * NHARM * 4);
  float* pn   = (float*)alloc((size_t)NTOK * NBANDS * 4);
  float* Cph  = (float*)alloc((size_t)NTOK * 4);
  float* sig  = (float*)alloc((size_t)NBATCH * LSIG * 4);
  float* imp  = (float*)alloc((size_t)LSIG * 4);

  // ---- weight prep ----
  k_w_transpose_pad<<<ceil_div(512 * 512, 256), 256, 0, stream>>>(W2t, iW[1], 512, 512, 512, 512);
  k_w_transpose_pad<<<ceil_div(512 * 512, 256), 256, 0, stream>>>(W3t, iW[2], 512, 512, 512, 512);
  k_w_transpose_pad<<<ceil_div(512 * 544, 256), 256, 0, stream>>>(O1t, oW[0], 514, 512, 544, 512);
  k_w_transpose_pad<<<ceil_div(512 * 512, 256), 256, 0, stream>>>(O2t, oW[1], 512, 512, 512, 512);
  k_w_transpose_pad<<<ceil_div(512 * 512, 256), 256, 0, stream>>>(O3t, oW[2], 512, 512, 512, 512);
  k_w_transpose_pad<<<ceil_div(128 * 512, 256), 256, 0, stream>>>(Wht, Wh, 512, NHARM + 1, 512, 128);
  k_w_transpose_pad<<<ceil_div(128 * 512, 256), 256, 0, stream>>>(Wnt, Wn, 512, NBANDS, 512, 128);
  k_w_convert<<<ceil_div(1536 * 1024, 256), 256, 0, stream>>>(WihC, Wih, 1536 * 1024);
  k_w_convert<<<ceil_div(1536 * 512, 256), 256, 0, stream>>>(WhhC, Whh, 1536 * 512);
  k_fill_f16<<<ceil_div((long)NTOK * 544, 256), 256, 0, stream>>>(X1, NTOK * 544);

  // ---- in_mlp on pitch -> Xcat[:,0:512]; on loudness -> Xcat[:,512:1024] (shared weights)
  const float* ins[2] = { pitch, loud };
  for (int p = 0; p < 2; ++p) {
    k_outer<<<ceil_div((long)NTOK * 512, 256), 256, 0, stream>>>(Y, ins[p], iW[0], ib[0], NTOK);
    k_ln_leaky<<<NTOK, 256, 0, stream>>>(Xa, Y, ig[0], ibe[0], 512, 0);
    launch_gemm(stream, Y, Xa, W2t, ib[1], NTOK, 512, 512, 512);
    k_ln_leaky<<<NTOK, 256, 0, stream>>>(Xb, Y, ig[1], ibe[1], 512, 0);
    launch_gemm(stream, Y, Xb, W3t, ib[2], NTOK, 512, 512, 512);
    k_ln_leaky<<<NTOK, 256, 0, stream>>>(Xcat, Y, ig[2], ibe[2], 1024, p * 512);
  }

  // ---- GRU: hoisted input GEMM, then persistent recurrent kernel -> X1[:,0:512]
  launch_gemm(stream, Gi, Xcat, WihC, bih, NTOK, 1536, 1536, 1024);
  k_gru<<<1, 512, 0, stream>>>(X1, Gi, WhhC, bhh);
  k_set_pl<<<ceil_div(NTOK, 256), 256, 0, stream>>>(X1, pitch, loud);

  // ---- out_mlp + heads ----
  launch_gemm(stream, Y, X1, O1t, ob[0], NTOK, 512, 512, 544);
  k_ln_leaky<<<NTOK, 256, 0, stream>>>(Xa, Y, og[0], obe[0], 512, 0);
  launch_gemm(stream, Y, Xa, O2t, ob[1], NTOK, 512, 512, 512);
  k_ln_leaky<<<NTOK, 256, 0, stream>>>(Xb, Y, og[1], obe[1], 512, 0);
  launch_gemm(stream, Y, Xb, O3t, ob[2], NTOK, 512, 512, 512);
  k_ln_leaky<<<NTOK, 256, 0, stream>>>(Xa, Y, og[2], obe[2], 512, 0);
  launch_gemm(stream, Yh, Xa, Wht, bh, NTOK, NHARM + 1, 128, 512);
  launch_gemm(stream, Yn, Xa, Wnt, bn, NTOK, NBANDS, 128, 512);

  // ---- synthesis ----
  k_harm_amps<<<NTOK, 128, 0, stream>>>(amps, Yh, pitch);
  k_noise_scale<<<ceil_div((long)NTOK * NBANDS, 256), 256, 0, stream>>>(pn, Yn, NTOK * NBANDS);
  k_phase<<<1, 32, 0, stream>>>(Cph, pitch);
  k_harmonic<<<NTOK, 160, 0, stream>>>(sig, amps, pitch, Cph);
  k_noise<<<NTOK, 160, 0, stream>>>(sig, pn, noise_in);

  // ---- reverb (Toeplitz WMMA convolution, 4 t-tiles per wave, branch-free) ----
  k_impulse<<<ceil_div(LSIG, 256), 256, 0, stream>>>(imp, rev_noise, rev_decay, rev_wet);
  k_prep_sigH<<<ceil_div((long)NBATCH * (LSIG + 64), 256), 256, 0, stream>>>(sigH, sig);
  k_prep_R<<<ceil_div(LSIG + 128, 256), 256, 0, stream>>>(Rv, imp);
  k_reverb<<<ceil_div((long)(LSIG / 64) * 32, 256), 256, 0, stream>>>(out, sigH, Rv);
}
